// Layer_40596030882578
// MI455X (gfx1250) — matrix operand order repaired and checked by reading, not compiled, not run on hardware
//
#include <hip/hip_runtime.h>
#include <hip/hip_bf16.h>
#include <math.h>

// ---------------- problem constants ----------------
#define Bb   2
#define Ss   2048
#define Dd   1024
#define Hh   16
#define DHh  64
#define DFFf 4096
#define NNn  2048
#define NPp  64
#define KNn  16
#define KPp  16
#define BS   (Bb*Ss)   // 4096 tokens

typedef __bf16 bf16_t;
typedef bf16_t v8bf  __attribute__((ext_vector_type(8)));
typedef bf16_t v16bf __attribute__((ext_vector_type(16)));
typedef float  v8f   __attribute__((ext_vector_type(8)));

#define WMMA_BF16(a, b, c) \
  __builtin_amdgcn_wmma_f32_16x16x32_bf16(false, (a), false, (b), (short)0, (c), false, false)

// ---------- gfx1250 async global->LDS copy (ASYNCcnt-tracked DMA) ----------
#if defined(__has_builtin)
#if __has_builtin(__builtin_amdgcn_global_load_async_to_lds_b128)
#define HAVE_ASYNC_LDS 1
#endif
#endif
#ifndef HAVE_ASYNC_LDS
#define HAVE_ASYNC_LDS 0
#endif

typedef int v4i_t __attribute__((ext_vector_type(4)));
typedef __attribute__((address_space(1))) v4i_t* gvp_t;  // global int4*
typedef __attribute__((address_space(3))) v4i_t* lvp_t;  // LDS int4*

__device__ __forceinline__ void async_copy16(const bf16_t* g, bf16_t* l)
{
#if HAVE_ASYNC_LDS
  __builtin_amdgcn_global_load_async_to_lds_b128((gvp_t)g, (lvp_t)l, 0, 0);
#else
  *(v8bf*)l = *(const v8bf*)g;  // synchronous fallback
#endif
}

__device__ __forceinline__ void wait_async0()
{
#if HAVE_ASYNC_LDS
#if defined(__has_builtin) && __has_builtin(__builtin_amdgcn_s_wait_asynccnt)
  __builtin_amdgcn_s_wait_asynccnt(0);
#else
  asm volatile("s_wait_asynccnt 0x0" ::: "memory");
#endif
#endif
}

// ============================================================
// Generic WMMA GEMM: C[M,N] = A[M,K](bf16,row-major) * Bt[N,ldb](bf16,N-major)
// B column index = k & (ldb-1)  (lets us reuse a [N,1024] matrix for K=2048).
// The 64x32 B panel for the block is staged into LDS via async DMA with
// double buffering (DMA of slice k+32 overlaps WMMAs on slice k).
// mode 0: out_bf16 = acc + bias
// mode 1: out_f32  = acc
// mode 2: out_bf16 = gelu( (acc+bias) * sigmoid(pattern_gate) )
// mode 3: out_f32  = acc + bias + res
// ============================================================
__global__ __launch_bounds__(256)
void gemm_wmma(const bf16_t* __restrict__ A, const bf16_t* __restrict__ Bt,
               const float* __restrict__ bias, float* __restrict__ outf,
               bf16_t* __restrict__ outb, const float* __restrict__ res,
               const float* __restrict__ gate_tab, const int* __restrict__ pkidx,
               const float* __restrict__ pkw,
               int M, int N, int K, int ldb, int mode)
{
  __shared__ __align__(32) bf16_t bsh[2][64 * 32];

  const int tid  = threadIdx.x;
  const int wave = tid >> 5;
  const int lane = tid & 31;
  const int l    = lane & 15;
  const int hi   = lane >> 4;
  const int wm   = wave & 3;      // 4 waves along M
  const int wn   = wave >> 2;     // 2 waves along N
  const int m0   = blockIdx.y * 64 + wm * 16;
  const int n0b  = blockIdx.x * 64;
  const int kmask = ldb - 1;

  // B-panel staging: thread copies 16B: panel row rr (0..63), chunk cc (0..3)
  const int rr = tid >> 2, cc = tid & 3;
  const bf16_t* bgrow = Bt + (size_t)(n0b + rr) * ldb;
  bf16_t* bdst[2] = { &bsh[0][rr * 32 + cc * 8], &bsh[1][rr * 32 + cc * 8] };

  v8f acc0 = {}; v8f acc1 = {};
  const bf16_t* arow = A + (size_t)(m0 + l) * K;
  // per-wave B fragment base offsets inside the LDS panel
  const int bo0 = (wn * 32 + l)      * 32 + hi * 16;
  const int bo1 = (wn * 32 + 16 + l) * 32 + hi * 16;

  // prologue: start DMA of first slice
  async_copy16(bgrow + ((0 + cc * 8) & kmask), bdst[0]);

  for (int k = 0; k < K; k += 32) {
    const int cur = (k >> 5) & 1;
    wait_async0();        // own async copies (slice cur) have landed in LDS
    __syncthreads();      // all waves' copies visible; prev reads of cur^1 done
    if (k + 32 < K)       // overlap next slice's DMA with this slice's WMMAs
      async_copy16(bgrow + ((k + 32 + cc * 8) & kmask), bdst[cur ^ 1]);

    // A fragment (16x32 bf16): lane<16 -> K {0..15}, lane>=16 -> {16..31}
    const bf16_t* ap = arow + k + hi * 16;
    v8bf alo = *(const v8bf*)ap;
    v8bf ahi = *(const v8bf*)(ap + 8);
    v16bf a;
#pragma unroll
    for (int i = 0; i < 8; i++) { a[i] = alo[i]; a[8 + i] = ahi[i]; }
    // B fragments from the LDS panel
    v16bf b0 = *(const v16bf*)(&bsh[cur][bo0]);
    v16bf b1 = *(const v16bf*)(&bsh[cur][bo1]);
    acc0 = WMMA_BF16(a, b0, acc0);
    acc1 = WMMA_BF16(a, b1, acc1);
  }

#pragma unroll
  for (int r = 0; r < 8; r++) {
    const int row = m0 + r + hi * 8;
    const int c0  = n0b + wn * 32 + l;
    const int c1  = n0b + wn * 32 + 16 + l;
    float v0 = acc0[r], v1 = acc1[r];
    if (mode == 0) {
      v0 += bias[c0]; v1 += bias[c1];
      outb[(size_t)row * N + c0] = (bf16_t)v0;
      outb[(size_t)row * N + c1] = (bf16_t)v1;
    } else if (mode == 1) {
      outf[(size_t)row * N + c0] = v0;
      outf[(size_t)row * N + c1] = v1;
    } else if (mode == 2) {
      float g0 = 0.f, g1 = 0.f;
#pragma unroll
      for (int j = 0; j < KPp; j++) {
        const int   pid = pkidx[row * KPp + j];
        const float w   = pkw  [row * KPp + j];
        g0 += w * gate_tab[(size_t)pid * DFFf + c0];
        g1 += w * gate_tab[(size_t)pid * DFFf + c1];
      }
      v0 += bias[c0]; v1 += bias[c1];
      v0 *= 1.f / (1.f + __expf(-g0));
      v1 *= 1.f / (1.f + __expf(-g1));
      v0 = 0.5f * v0 * (1.f + erff(v0 * 0.70710678118654752f));
      v1 = 0.5f * v1 * (1.f + erff(v1 * 0.70710678118654752f));
      outb[(size_t)row * N + c0] = (bf16_t)v0;
      outb[(size_t)row * N + c1] = (bf16_t)v1;
    } else { // mode 3
      v0 += bias[c0] + res[(size_t)row * N + c0];
      v1 += bias[c1] + res[(size_t)row * N + c1];
      outf[(size_t)row * N + c0] = v0;
      outf[(size_t)row * N + c1] = v1;
    }
  }
}

// ============================================================
// Flash attention: one wave per 16-query tile, loop keys in tiles of 32.
// q,k: bf16 [B*S, D] (head-major inner dh).  vT: bf16 [B*H, DH, S].
// ctx: fp32 [B,S,H,DH].
// ============================================================
__global__ __launch_bounds__(128)
void attn_wmma(const bf16_t* __restrict__ qb, const bf16_t* __restrict__ kb,
               const bf16_t* __restrict__ vT, float* __restrict__ ctx)
{
  __shared__ __align__(32) bf16_t pshared[4][16 * 32];
  const int tid  = threadIdx.x;
  const int wave = tid >> 5;
  const int lane = tid & 31;
  const int l    = lane & 15;
  const int hi   = lane >> 4;
  const int bh = blockIdx.y;
  const int b  = bh >> 4;
  const int h  = bh & 15;
  const int q0 = (blockIdx.x * 4 + wave) * 16;

  // Q fragments (16x64 -> two 16x32 A-frags)
  const bf16_t* qrow = qb + (size_t)(b * Ss + q0 + l) * Dd + h * DHh;
  v16bf aq0, aq1;
  {
    v8bf x0 = *(const v8bf*)(qrow + hi * 16);
    v8bf x1 = *(const v8bf*)(qrow + 8 + hi * 16);
    v8bf x2 = *(const v8bf*)(qrow + 32 + hi * 16);
    v8bf x3 = *(const v8bf*)(qrow + 40 + hi * 16);
#pragma unroll
    for (int i = 0; i < 8; i++) {
      aq0[i] = x0[i]; aq0[8 + i] = x1[i];
      aq1[i] = x2[i]; aq1[8 + i] = x3[i];
    }
  }

  v8f o0 = {}, o1 = {}, o2 = {}, o3 = {};
  float mrow[8], lrow[8];
#pragma unroll
  for (int r = 0; r < 8; r++) { mrow[r] = -1e30f; lrow[r] = 0.f; }
  const float scale = 0.125f; // 1/sqrt(64)

  for (int kk = 0; kk < Ss; kk += 32) {
    v8f s0 = {}, s1 = {};
    // K^T B-frags: per-lane column = key, contiguous dh chunk at hi*16
    const bf16_t* k0p = kb + (size_t)(b * Ss + kk + l)      * Dd + h * DHh + hi * 16;
    const bf16_t* k1p = kb + (size_t)(b * Ss + kk + 16 + l) * Dd + h * DHh + hi * 16;
    v16bf bk00 = *(const v16bf*)(k0p);
    v16bf bk01 = *(const v16bf*)(k0p + 32);
    v16bf bk10 = *(const v16bf*)(k1p);
    v16bf bk11 = *(const v16bf*)(k1p + 32);
    s0 = WMMA_BF16(aq0, bk00, s0);
    s0 = WMMA_BF16(aq1, bk01, s0);
    s1 = WMMA_BF16(aq0, bk10, s1);
    s1 = WMMA_BF16(aq1, bk11, s1);

    // online softmax update per row (rows live per vgpr r, replicated across 16 lanes)
    float alpha[8];
#pragma unroll
    for (int r = 0; r < 8; r++) {
      float v0 = s0[r] * scale, v1 = s1[r] * scale;
      float mx = fmaxf(v0, v1);
      mx = fmaxf(mx, __shfl_xor(mx, 1, 32));
      mx = fmaxf(mx, __shfl_xor(mx, 2, 32));
      mx = fmaxf(mx, __shfl_xor(mx, 4, 32));
      mx = fmaxf(mx, __shfl_xor(mx, 8, 32));
      const float mnew = fmaxf(mrow[r], mx);
      const float a = __expf(mrow[r] - mnew);
      const float p0 = __expf(v0 - mnew);
      const float p1 = __expf(v1 - mnew);
      float ps = p0 + p1;
      ps += __shfl_xor(ps, 1, 32);
      ps += __shfl_xor(ps, 2, 32);
      ps += __shfl_xor(ps, 4, 32);
      ps += __shfl_xor(ps, 8, 32);
      lrow[r] = lrow[r] * a + ps;
      mrow[r] = mnew;
      alpha[r] = a;
      const int mloc = r + hi * 8;
      pshared[wave][mloc * 32 + l]      = (bf16_t)p0;
      pshared[wave][mloc * 32 + 16 + l] = (bf16_t)p1;
    }
#pragma unroll
    for (int r = 0; r < 8; r++) {
      o0[r] *= alpha[r]; o1[r] *= alpha[r];
      o2[r] *= alpha[r]; o3[r] *= alpha[r];
    }
    // re-read P as an A-fragment (DS ops are in-order within the wave)
    v16bf apf;
    {
      const bf16_t* pp = &pshared[wave][l * 32 + hi * 16];
      v8bf plo = *(const v8bf*)pp;
      v8bf phi = *(const v8bf*)(pp + 8);
#pragma unroll
      for (int i = 0; i < 8; i++) { apf[i] = plo[i]; apf[8 + i] = phi[i]; }
    }
    // V B-frags from transposed V: per-lane column dh, contiguous keys at hi*16
    const bf16_t* vbase = vT + (size_t)bh * DHh * Ss + kk + hi * 16;
    v16bf bv0 = *(const v16bf*)(vbase + (size_t)(0 * 16 + l) * Ss);
    v16bf bv1 = *(const v16bf*)(vbase + (size_t)(1 * 16 + l) * Ss);
    v16bf bv2 = *(const v16bf*)(vbase + (size_t)(2 * 16 + l) * Ss);
    v16bf bv3 = *(const v16bf*)(vbase + (size_t)(3 * 16 + l) * Ss);
    o0 = WMMA_BF16(apf, bv0, o0);
    o1 = WMMA_BF16(apf, bv1, o1);
    o2 = WMMA_BF16(apf, bv2, o2);
    o3 = WMMA_BF16(apf, bv3, o3);
  }

#pragma unroll
  for (int r = 0; r < 8; r++) {
    const float inv = 1.f / lrow[r];
    const size_t base = ((size_t)(b * Ss + q0 + r + hi * 8) * Hh + h) * DHh;
    ctx[base + 0 * 16 + l] = o0[r] * inv;
    ctx[base + 1 * 16 + l] = o1[r] * inv;
    ctx[base + 2 * 16 + l] = o2[r] * inv;
    ctx[base + 3 * 16 + l] = o3[r] * inv;
  }
}

// ============================================================
// small helper kernels
// ============================================================
__global__ __launch_bounds__(256)
void ln_kernel(const float* __restrict__ x, const float* __restrict__ g,
               const float* __restrict__ bta, bf16_t* __restrict__ outb)
{
  __shared__ float rs[256], rq[256];
  const int t = blockIdx.x, tid = threadIdx.x;
  const float* row = x + (size_t)t * Dd;
  float s = 0.f, q = 0.f;
  for (int d = tid; d < Dd; d += 256) { float v = row[d]; s += v; q += v * v; }
  rs[tid] = s; rq[tid] = q; __syncthreads();
  for (int st = 128; st > 0; st >>= 1) {
    if (tid < st) { rs[tid] += rs[tid + st]; rq[tid] += rq[tid + st]; }
    __syncthreads();
  }
  const float mean = rs[0] * (1.f / Dd);
  const float var  = rq[0] * (1.f / Dd) - mean * mean;
  const float inv  = rsqrtf(var + 1e-5f);
  for (int d = tid; d < Dd; d += 256) {
    const float v = (row[d] - mean) * inv * g[d] + bta[d];
    outb[(size_t)t * Dd + d] = (bf16_t)v;
  }
}

__global__ __launch_bounds__(256)
void transpose_bf16_kernel(const float* __restrict__ W, bf16_t* __restrict__ out,
                           int K, int N)
{
  __shared__ float tile[32][33];
  const int kb = blockIdx.y * 32, nb = blockIdx.x * 32;
  const int tx = threadIdx.x & 31, ty = threadIdx.x >> 5;
#pragma unroll
  for (int i = 0; i < 32; i += 8)
    tile[ty + i][tx] = W[(size_t)(kb + ty + i) * N + nb + tx];
  __syncthreads();
#pragma unroll
  for (int i = 0; i < 32; i += 8)
    out[(size_t)(nb + ty + i) * K + kb + tx] = (bf16_t)tile[tx][ty + i];
}

__global__ __launch_bounds__(256)
void cvt_bf16_kernel(const float* __restrict__ src, bf16_t* __restrict__ dst, int n)
{
  const int i = blockIdx.x * 256 + threadIdx.x;
  if (i < n) dst[i] = (bf16_t)src[i];
}

__global__ __launch_bounds__(256)
void vtrans_kernel(const bf16_t* __restrict__ vb, bf16_t* __restrict__ vT)
{
  const size_t o = (size_t)blockIdx.x * 256 + threadIdx.x; // [B*H, DH, S]
  const int s = (int)(o % Ss);
  const size_t r = o / Ss;
  const int d = (int)(r % DHh);
  const int bh = (int)(r / DHh);
  const int b = bh >> 4, h = bh & 15;
  vT[o] = vb[(size_t)(b * Ss + s) * Dd + h * DHh + d];
}

__global__ __launch_bounds__(256)
void path_kernel(const bf16_t* __restrict__ normedb, const float* __restrict__ ctx,
                 const float* __restrict__ Wpath, const float* __restrict__ bpath,
                 float* __restrict__ pathw)
{
  __shared__ float r0[256], r1[256];
  const int t = blockIdx.x, tid = threadIdx.x;
  float s0 = 0.f, s1 = 0.f;
  for (int d = tid; d < 2 * Dd; d += 256) {
    const float v = (d < Dd) ? (float)normedb[(size_t)t * Dd + d]
                             : ctx[(size_t)t * Dd + d - Dd];
    s0 += v * Wpath[d * 2 + 0];
    s1 += v * Wpath[d * 2 + 1];
  }
  r0[tid] = s0; r1[tid] = s1; __syncthreads();
  for (int st = 128; st > 0; st >>= 1) {
    if (tid < st) { r0[tid] += r0[tid + st]; r1[tid] += r1[tid + st]; }
    __syncthreads();
  }
  if (tid == 0) {
    const float a = r0[0] + bpath[0], b = r1[0] + bpath[1];
    const float mx = fmaxf(a, b);
    const float e0 = __expf(a - mx), e1 = __expf(b - mx);
    const float inv = 1.f / (e0 + e1);
    pathw[t * 2 + 0] = e0 * inv;
    pathw[t * 2 + 1] = e1 * inv;
  }
}

__global__ __launch_bounds__(256)
void buildA_kernel(const bf16_t* __restrict__ normedb, const float* __restrict__ ctx,
                   const float* __restrict__ pathw, bf16_t* __restrict__ Ap)
{
  const int t = blockIdx.x, tid = threadIdx.x;
  const float w0 = pathw[t * 2], w1 = pathw[t * 2 + 1];
  for (int d = tid; d < 2 * Dd; d += 256) {
    const float v = (d < Dd) ? w0 * (float)normedb[(size_t)t * Dd + d]
                             : w1 * ctx[(size_t)t * Dd + d - Dd];
    Ap[(size_t)t * (2 * Dd) + d] = (bf16_t)v;
  }
}

__global__ __launch_bounds__(256)
void route_topk_kernel(const float* __restrict__ scores, const float* __restrict__ neurons,
                       const float* __restrict__ x, const float* __restrict__ aff,
                       float* __restrict__ x2, int* __restrict__ out_idx,
                       float* __restrict__ out_sel, int* __restrict__ pk_idx,
                       float* __restrict__ pk_w)
{
  __shared__ float sc[NNn];
  __shared__ float rmax[256];
  __shared__ int   rarg[256];
  __shared__ int   tki[KNn];
  __shared__ float tkw[KNn];
  __shared__ float ps[NPp];
  const int t = blockIdx.x, tid = threadIdx.x;
  for (int n = tid; n < NNn; n += 256) sc[n] = scores[(size_t)t * NNn + n];
  __syncthreads();
  for (int it = 0; it < KNn; it++) {
    float best = -1e30f; int bi = 0x7fffffff;
    for (int n = tid; n < NNn; n += 256) {
      const float v = sc[n];
      if (v > best || (v == best && n < bi)) { best = v; bi = n; }
    }
    rmax[tid] = best; rarg[tid] = bi; __syncthreads();
    for (int st = 128; st > 0; st >>= 1) {
      if (tid < st) {
        if (rmax[tid + st] > rmax[tid] ||
            (rmax[tid + st] == rmax[tid] && rarg[tid + st] < rarg[tid])) {
          rmax[tid] = rmax[tid + st]; rarg[tid] = rarg[tid + st];
        }
      }
      __syncthreads();
    }
    if (tid == 0) { tki[it] = rarg[0]; tkw[it] = rmax[0]; sc[rarg[0]] = -1e30f; }
    __syncthreads();
  }
  if (tid == 0) { // softmax over sorted-desc top-k scores
    const float mx = tkw[0];
    float s = 0.f;
    for (int i = 0; i < KNn; i++) { tkw[i] = __expf(tkw[i] - mx); s += tkw[i]; }
    const float inv = 1.f / s;
    for (int i = 0; i < KNn; i++) tkw[i] *= inv;
  }
  __syncthreads();
  if (tid < KNn) out_idx[t * KNn + tid] = tki[tid];
  for (int n = tid; n < NNn; n += 256) out_sel[(size_t)t * NNn + n] = 0.f;
  __syncthreads();
  if (tid < KNn) out_sel[(size_t)t * NNn + tki[tid]] = tkw[tid];
  // router out + residual
  for (int c = tid; c < Dd; c += 256) {
    float a = 0.f;
#pragma unroll
    for (int j = 0; j < KNn; j++) a += tkw[j] * neurons[(size_t)tki[j] * Dd + c];
    x2[(size_t)t * Dd + c] = x[(size_t)t * Dd + c] + a;
  }
  // pattern scores, top-KP, softmax
  if (tid < NPp) {
    float a = 0.f;
#pragma unroll
    for (int j = 0; j < KNn; j++) a += tkw[j] * aff[(size_t)tid * NNn + tki[j]];
    ps[tid] = a;
  }
  __syncthreads();
  if (tid == 0) {
    float vals[KPp]; int ids[KPp];
    for (int it = 0; it < KPp; it++) {
      float best = -1e30f; int bi = 0;
      for (int p = 0; p < NPp; p++) if (ps[p] > best) { best = ps[p]; bi = p; }
      vals[it] = best; ids[it] = bi; ps[bi] = -1e30f;
    }
    const float mx = vals[0];
    float s = 0.f;
    for (int i = 0; i < KPp; i++) { vals[i] = __expf(vals[i] - mx); s += vals[i]; }
    const float inv = 1.f / s;
    for (int i = 0; i < KPp; i++) { pk_idx[t * KPp + i] = ids[i]; pk_w[t * KPp + i] = vals[i] * inv; }
  }
}

// ============================================================
// host launcher
// ============================================================
extern "C" void kernel_launch(void* const* d_in, const int* in_sizes, int n_in,
                              void* d_out, int out_size, void* d_ws, size_t ws_size,
                              hipStream_t stream)
{
  (void)in_sizes; (void)n_in; (void)out_size; (void)ws_size;
  const float* x       = (const float*)d_in[0];
  const float* neurons = (const float*)d_in[1];
  const float* Wq      = (const float*)d_in[2];
  const float* bq      = (const float*)d_in[3];
  const float* Wk      = (const float*)d_in[4];
  const float* bk      = (const float*)d_in[5];
  const float* Wv      = (const float*)d_in[6];
  const float* bv      = (const float*)d_in[7];
  const float* Wpath   = (const float*)d_in[8];
  const float* bpath   = (const float*)d_in[9];
  const float* aff     = (const float*)d_in[10];
  const float* gates   = (const float*)d_in[11];
  const float* Wup     = (const float*)d_in[12];
  const float* bup     = (const float*)d_in[13];
  const float* Wdown   = (const float*)d_in[14];
  const float* bdown   = (const float*)d_in[15];
  const float* ln1g    = (const float*)d_in[16];
  const float* ln1b    = (const float*)d_in[17];
  const float* ln2g    = (const float*)d_in[18];
  const float* ln2b    = (const float*)d_in[19];

  float* out_x   = (float*)d_out;                    // [B,S,D]
  int*   out_idx = (int*)(out_x + (size_t)BS * Dd);  // [B,S,KN] (int bits)
  float* out_sel = out_x + (size_t)BS * Dd + (size_t)BS * KNn; // [B,S,NN]

  char* w = (char*)d_ws;
  size_t off = 0;
  auto alloc = [&](size_t bytes) -> void* {
    void* p = w + off;
    off = (off + bytes + 255) & ~(size_t)255;
    return p;
  };
  bf16_t* normedb  = (bf16_t*)alloc((size_t)BS * Dd * 2);
  bf16_t* WqT      = (bf16_t*)alloc((size_t)Dd * Dd * 2);
  bf16_t* WkT      = (bf16_t*)alloc((size_t)Dd * Dd * 2);
  bf16_t* WvT      = (bf16_t*)alloc((size_t)Dd * Dd * 2);
  bf16_t* neuronsb = (bf16_t*)alloc((size_t)NNn * Dd * 2);
  bf16_t* qb       = (bf16_t*)alloc((size_t)BS * Dd * 2);
  bf16_t* kbuf     = (bf16_t*)alloc((size_t)BS * Dd * 2);
  bf16_t* vb       = (bf16_t*)alloc((size_t)BS * Dd * 2);
  bf16_t* vTb      = (bf16_t*)alloc((size_t)BS * Dd * 2);
  float*  ctx      = (float*) alloc((size_t)BS * Dd * 4);
  float*  pathw    = (float*) alloc((size_t)BS * 2 * 4);
  bf16_t* Ap       = (bf16_t*)alloc((size_t)BS * 2 * Dd * 2);
  float*  scores   = (float*) alloc((size_t)BS * NNn * 4);
  float*  x2       = (float*) alloc((size_t)BS * Dd * 4);
  bf16_t* normed2b = (bf16_t*)alloc((size_t)BS * Dd * 2);
  int*    pkidx    = (int*)   alloc((size_t)BS * KPp * 4);
  float*  pkw      = (float*) alloc((size_t)BS * KPp * 4);
  bf16_t* WupT     = (bf16_t*)alloc((size_t)DFFf * Dd * 2);
  bf16_t* hb       = (bf16_t*)alloc((size_t)BS * DFFf * 2);
  bf16_t* WdownT   = (bf16_t*)alloc((size_t)Dd * DFFf * 2);

  // --- weight precondition (transpose + bf16) ---
  transpose_bf16_kernel<<<dim3(Dd/32, Dd/32), 256, 0, stream>>>(Wq, WqT, Dd, Dd);
  transpose_bf16_kernel<<<dim3(Dd/32, Dd/32), 256, 0, stream>>>(Wk, WkT, Dd, Dd);
  transpose_bf16_kernel<<<dim3(Dd/32, Dd/32), 256, 0, stream>>>(Wv, WvT, Dd, Dd);
  transpose_bf16_kernel<<<dim3(DFFf/32, Dd/32), 256, 0, stream>>>(Wup, WupT, Dd, DFFf);
  transpose_bf16_kernel<<<dim3(Dd/32, DFFf/32), 256, 0, stream>>>(Wdown, WdownT, DFFf, Dd);
  cvt_bf16_kernel<<<(NNn * Dd) / 256, 256, 0, stream>>>(neurons, neuronsb, NNn * Dd);

  // --- LN1 ---
  ln_kernel<<<BS, 256, 0, stream>>>(x, ln1g, ln1b, normedb);

  // --- QKV projections (WMMA bf16, async-LDS staged B) ---
  const dim3 gq(Dd / 64, BS / 64);
  gemm_wmma<<<gq, 256, 0, stream>>>(normedb, WqT, bq, nullptr, qb,   nullptr, nullptr, nullptr, nullptr, BS, Dd, Dd, Dd, 0);
  gemm_wmma<<<gq, 256, 0, stream>>>(normedb, WkT, bk, nullptr, kbuf, nullptr, nullptr, nullptr, nullptr, BS, Dd, Dd, Dd, 0);
  gemm_wmma<<<gq, 256, 0, stream>>>(normedb, WvT, bv, nullptr, vb,   nullptr, nullptr, nullptr, nullptr, BS, Dd, Dd, Dd, 0);
  vtrans_kernel<<<(BS * Dd) / 256, 256, 0, stream>>>(vb, vTb);

  // --- flash attention ---
  attn_wmma<<<dim3(Ss / 64, Bb * Hh), 128, 0, stream>>>(qb, kbuf, vTb, ctx);

  // --- path weights + fused blended neuron-score GEMM ---
  path_kernel<<<BS, 256, 0, stream>>>(normedb, ctx, Wpath, bpath, pathw);
  buildA_kernel<<<BS, 256, 0, stream>>>(normedb, ctx, pathw, Ap);
  gemm_wmma<<<dim3(NNn / 64, BS / 64), 256, 0, stream>>>(Ap, neuronsb, nullptr, scores, nullptr,
                                                         nullptr, nullptr, nullptr, nullptr,
                                                         BS, NNn, 2 * Dd, Dd, 1);

  // --- top-k routing, residual, pattern selection ---
  route_topk_kernel<<<BS, 256, 0, stream>>>(scores, neurons, x, aff,
                                            x2, out_idx, out_sel, pkidx, pkw);

  // --- LN2 + gated FFN ---
  ln_kernel<<<BS, 256, 0, stream>>>(x2, ln2g, ln2b, normed2b);
  gemm_wmma<<<dim3(DFFf / 64, BS / 64), 256, 0, stream>>>(normed2b, WupT, bup, nullptr, hb,
                                                          nullptr, gates, pkidx, pkw,
                                                          BS, DFFf, Dd, Dd, 2);
  gemm_wmma<<<dim3(Dd / 64, BS / 64), 256, 0, stream>>>(hb, WdownT, bdown, out_x, nullptr,
                                                        x2, nullptr, nullptr, nullptr,
                                                        BS, Dd, DFFf, DFFf, 3);
}
